// ForcedStepper_62543313764867
// MI455X (gfx1250) — matrix operand order repaired and checked by reading, not compiled, not run on hardware
//
#include <hip/hip_runtime.h>
#include <hip/hip_bf16.h>

// ---------------------------------------------------------------------------
// Problem constants (from reference)
// ---------------------------------------------------------------------------
#define T_   32
#define B_   8192
#define Z_   34
#define M_   68          // 2*Z
#define N2D_ 3
#define NF_  71          // N2D + M  (BN features)
#define DIN_ 139         // M + N2D + M
#define KP_  160         // DIN padded to multiple of 32
#define H_   256
#define NP_  80          // M padded to multiple of 16
#define NSTEPS_ 2
#define DTSTEP_ 5400.0f  // DT / NSTEPS
#define EPSQ_ 1e-9f
#define BNEPS_ 1e-5f
#define LV_ 2510000.0f

#define PROG_CNT (T_ * B_ * M_)          // 17825792 floats
#define QNN_SCALE (1.0f / (DTSTEP_ * 86400.0f * 1.0e6f * (float)NSTEPS_))
#define SNN_SCALE (1004.0f / (DTSTEP_ * 86400.0f * (float)NSTEPS_))

typedef __attribute__((ext_vector_type(16))) _Float16 v16h;
typedef __attribute__((ext_vector_type(8)))  _Float16 v8h;
typedef __attribute__((ext_vector_type(8)))  float    v8f;

// Wave-local fence: per-wave LDS tiles are private and CDNA5 LDS ops from one
// wave are processed in order, so cross-lane RAW through LDS needs only a
// compiler reordering fence, not a hardware barrier.
__device__ __forceinline__ void wave_sync() {
    asm volatile("" ::: "memory");
    __builtin_amdgcn_wave_barrier();
}

// ---------------------------------------------------------------------------
// Raw (un-normalized) BN feature value for (t, b, f), f in [0,71)
// d2 = [shf, lhf, solin, forc_sl(34), forc_qt(34)]
// ---------------------------------------------------------------------------
__device__ __forceinline__ float raw_feat(int t, int b, int f,
                                          const float* __restrict__ shf,
                                          const float* __restrict__ lhf,
                                          const float* __restrict__ solin,
                                          const float* __restrict__ fsl,
                                          const float* __restrict__ fqt) {
    int tb = t * B_ + b;
    if (f == 0) return shf[tb];
    if (f == 1) return lhf[tb];
    if (f == 2) return solin[tb];
    if (f < 37) return fsl[tb * Z_ + (f - 3)];
    return fqt[tb * Z_ + (f - 37)];
}

// ---------------------------------------------------------------------------
// Pass 1: BN statistics per (t, feature).  Hoisted out of the time loop
// because d2 depends only on forcing.  Writes scale/shift pairs to ws.
// ---------------------------------------------------------------------------
__global__ __launch_bounds__(256) void bn_stats_kernel(
    const float* __restrict__ shf, const float* __restrict__ lhf,
    const float* __restrict__ solin, const float* __restrict__ fsl,
    const float* __restrict__ fqt, const float* __restrict__ gamma,
    const float* __restrict__ beta, float* __restrict__ stats) {
    const int t = blockIdx.x / NF_;
    const int f = blockIdx.x % NF_;
    __shared__ float red1[256];
    __shared__ float red2[256];
    float s = 0.f, sq = 0.f;
    for (int b = threadIdx.x; b < B_; b += 256) {
        float v = raw_feat(t, b, f, shf, lhf, solin, fsl, fqt);
        s += v; sq += v * v;
    }
    red1[threadIdx.x] = s;
    red2[threadIdx.x] = sq;
    __syncthreads();
    for (int off = 128; off > 0; off >>= 1) {
        if (threadIdx.x < off) {
            red1[threadIdx.x] += red1[threadIdx.x + off];
            red2[threadIdx.x] += red2[threadIdx.x + off];
        }
        __syncthreads();
    }
    if (threadIdx.x == 0) {
        float mu  = red1[0] * (1.0f / (float)B_);
        float var = red2[0] * (1.0f / (float)B_) - mu * mu;
        float sc  = gamma[f] * rsqrtf(var + BNEPS_);
        stats[(t * NF_ + f) * 2 + 0] = sc;
        stats[(t * NF_ + f) * 2 + 1] = beta[f] - mu * sc;
    }
}

// ---------------------------------------------------------------------------
// Pass 2: persistent time-stepper.  4 waves per block share the f16 weight
// tiles in LDS; each wave owns an independent 16-column tile and runs all
// 31x2 sequential steps with only wave-local fences.  The two MLP layers run
// on v_wmma_f32_16x16x32_f16; state/physics in fp32.
// ---------------------------------------------------------------------------
__global__ __launch_bounds__(128) void forced_stepper_kernel(
    const float* __restrict__ prog_sl, const float* __restrict__ prog_qt,
    const float* __restrict__ forc_sl, const float* __restrict__ forc_qt,
    const float* __restrict__ shf, const float* __restrict__ lhf,
    const float* __restrict__ solin, const float* __restrict__ w,
    const float* __restrict__ W1, const float* __restrict__ b1,
    const float* __restrict__ W2, const float* __restrict__ b2,
    const float* __restrict__ stats, float* __restrict__ out) {

    // ---- LDS (~218 KB total; weights shared by all 4 waves) ----
    __shared__ __align__(16) _Float16 W1L[H_ * KP_];      // [n][k] 80 KB
    __shared__ __align__(16) _Float16 W2L[NP_ * H_];      // [n][k] 40 KB
    __shared__ __align__(16) _Float16 XL[4][16 * KP_];    // per-wave activations
    __shared__ __align__(16) _Float16 HL[4][16 * H_];     // per-wave hidden
    __shared__ __align__(16) float    YL[4][16 * NP_];    // per-wave MLP out (f32)
    __shared__ float SLs[4][16 * Z_];
    __shared__ float QTs[4][16 * Z_];
    __shared__ float STATSL[4][NF_ * 2 + 2];
    __shared__ float B1L[H_];
    __shared__ float B2L[NP_];
    __shared__ float WL[Z_ + 2];

    const int tid  = threadIdx.x;
    const int wv   = tid >> 5;              // wave id 0..3
    const int lane = tid & 31;
    const int b0   = blockIdx.x * 64 + wv * 16;   // 128 blocks * 4 waves * 16 cols

    // ---- one-time cooperative staging (all 128 threads) ----
    for (int idx = tid; idx < H_ * KP_; idx += 128) {
        int n = idx / KP_, k = idx % KP_;
        W1L[idx] = (k < DIN_) ? (_Float16)W1[k * H_ + n] : (_Float16)0.0f;
    }
    for (int idx = tid; idx < NP_ * H_; idx += 128) {
        int n = idx / H_, k = idx % H_;
        W2L[idx] = (n < M_) ? (_Float16)W2[k * M_ + n] : (_Float16)0.0f;
    }
    for (int idx = tid; idx < H_; idx += 128) B1L[idx] = b1[idx];
    if (tid < NP_) B2L[tid] = (tid < M_) ? b2[tid] : 0.f;
    if (tid < Z_)  WL[tid]  = w[tid];

    // per-wave init: zero-pad X columns [DIN, KP), initial state, t=0 output
    _Float16* XLw = XL[wv];
    _Float16* HLw = HL[wv];
    float*    YLw = YL[wv];
    float*    SLw = SLs[wv];
    float*    QTw = QTs[wv];
    float*    STw = STATSL[wv];

    for (int idx = lane; idx < 16 * (KP_ - DIN_); idx += 32) {
        int c = idx / (KP_ - DIN_), kk = DIN_ + idx % (KP_ - DIN_);
        XLw[c * KP_ + kk] = (_Float16)0.0f;
    }
    for (int idx = lane; idx < 16 * Z_; idx += 32) {
        int c = idx / Z_, z = idx % Z_;
        float sl0 = prog_sl[(b0 + c) * Z_ + z];
        float qt0 = prog_qt[(b0 + c) * Z_ + z];
        SLw[c * Z_ + z] = sl0;
        QTw[c * Z_ + z] = qt0;
        out[(b0 + c) * M_ + z]      = sl0;   // prog slice t=0, sl part
        out[(b0 + c) * M_ + Z_ + z] = qt0;   // prog slice t=0, qt part
    }
    __syncthreads();   // weights visible to all 4 waves; waves decouple after this

    float cnorm2 = 0.f;
    for (int z = 0; z < Z_; ++z) { float cz = -WL[z] * 1e-3f; cnorm2 += cz * cz; }

    const int mrow  = lane & 15;
    const int koff  = (lane & 16) ? 8 : 0;   // A-fragment chunk base offset
    const int bk    = (lane & 16) ? 8 : 0;   // B-fragment chunk base offset
    const int mbase = (lane & 16) ? 8 : 0;   // C/D row base

    // ======================= sequential outer time loop =====================
    for (int t = 0; t < T_ - 1; ++t) {
        // prefetch next step's forcing rows into cache (global_prefetch_b8)
        if (t + 1 < T_ - 1) {
            int bpf = ((t + 1) * B_ + b0 + mrow) * Z_;
            __builtin_prefetch(&forc_sl[bpf], 0, 1);
            __builtin_prefetch(&forc_qt[bpf], 0, 1);
        }
        // per-t BN scale/shift into this wave's LDS slot
        for (int idx = lane; idx < NF_ * 2; idx += 32)
            STw[idx] = stats[t * NF_ * 2 + idx];
        wave_sync();
        // normalized forcing features -> X columns [68,139) (inner-step invariant)
        for (int idx = lane; idx < 16 * NF_; idx += 32) {
            int c = idx / NF_, f = idx % NF_;
            float rv = raw_feat(t, b0 + c, f, shf, lhf, solin, forc_sl, forc_qt);
            XLw[c * KP_ + (2 * Z_) + f] = (_Float16)(rv * STw[f * 2] + STw[f * 2 + 1]);
        }
        float evap = 0.f, qnn = 0.f, snn = 0.f;
        if (lane < 16) evap = lhf[t * B_ + (b0 + lane)] * (86400.0f / LV_);
        wave_sync();

        // ======================= inner sub-steps ============================
        for (int step = 0; step < NSTEPS_; ++step) {
            // state -> X columns [0,68)
            for (int idx = lane; idx < 16 * (2 * Z_); idx += 32) {
                int c = idx / (2 * Z_), f = idx % (2 * Z_);
                float v = (f < Z_) ? SLw[c * Z_ + f] : QTw[c * Z_ + (f - Z_)];
                XLw[c * KP_ + f] = (_Float16)v;
            }
            wave_sync();

            // ---------------- layer 1: (16xKP_) @ (KP_ x 256), relu --------
            v16h A1[KP_ / 32];
            #pragma unroll
            for (int s = 0; s < KP_ / 32; ++s) {
                union { v16h v; v8h h[2]; } u;
                const _Float16* p = &XLw[mrow * KP_ + s * 32 + koff];
                u.h[0] = *(const v8h*)(p);
                u.h[1] = *(const v8h*)(p + 16);
                A1[s] = u.v;
            }
            #pragma unroll
            for (int nt = 0; nt < H_ / 16; ++nt) {
                float bv = B1L[nt * 16 + mrow];
                v8f acc = {bv, bv, bv, bv, bv, bv, bv, bv};
                #pragma unroll
                for (int s = 0; s < KP_ / 32; ++s) {
                    union { v16h v; v8h h[2]; } ub;
                    const _Float16* p = &W1L[(nt * 16 + mrow) * KP_ + s * 32 + bk];
                    ub.h[0] = *(const v8h*)(p);
                    ub.h[1] = *(const v8h*)(p + 16);
                    acc = __builtin_amdgcn_wmma_f32_16x16x32_f16(
                        false, A1[s], false, ub.v, (short)0, acc, false, false);
                }
                #pragma unroll
                for (int r = 0; r < 8; ++r)
                    HLw[(mbase + r) * H_ + nt * 16 + mrow] =
                        (_Float16)fmaxf(acc[r], 0.0f);
            }
            wave_sync();

            // ---------------- layer 2: (16x256) @ (256 x NP_) --------------
            v16h A2[H_ / 32];
            #pragma unroll
            for (int s = 0; s < H_ / 32; ++s) {
                union { v16h v; v8h h[2]; } u;
                const _Float16* p = &HLw[mrow * H_ + s * 32 + koff];
                u.h[0] = *(const v8h*)(p);
                u.h[1] = *(const v8h*)(p + 16);
                A2[s] = u.v;
            }
            #pragma unroll
            for (int nt = 0; nt < NP_ / 16; ++nt) {
                float bv = B2L[nt * 16 + mrow];
                v8f acc = {bv, bv, bv, bv, bv, bv, bv, bv};
                #pragma unroll
                for (int s = 0; s < H_ / 32; ++s) {
                    union { v16h v; v8h h[2]; } ub;
                    const _Float16* p = &W2L[(nt * 16 + mrow) * H_ + s * 32 + bk];
                    ub.h[0] = *(const v8h*)(p);
                    ub.h[1] = *(const v8h*)(p + 16);
                    acc = __builtin_amdgcn_wmma_f32_16x16x32_f16(
                        false, A2[s], false, ub.v, (short)0, acc, false, false);
                }
                #pragma unroll
                for (int r = 0; r < 8; ++r)
                    YLw[(mbase + r) * NP_ + nt * 16 + mrow] = acc[r];
            }
            wave_sync();

            // ---------------- per-column physics (fp32) --------------------
            if (lane < 16) {
                const int c = lane;
                float q0 = 0.f, s0 = 0.f, linq = 0.f;
                for (int z = 0; z < Z_; ++z) {
                    q0   += QTw[c * Z_ + z] * WL[z];
                    s0   += SLw[c * Z_ + z] * WL[z];
                    linq += YLw[c * NP_ + Z_ + z] * WL[z];
                }
                linq = -linq * 1e-3f;                 // -mass_int(src_qt)/1000
                float a   = -evap;
                float add = (linq < a) ? (a - linq) / cnorm2 : 0.f;
                float total = 0.f, lack = 0.f, valid = 0.f;
                for (int z = 0; z < Z_; ++z) {
                    SLw[c * Z_ + z] += DTSTEP_ * YLw[c * NP_ + z];
                    float sqt = YLw[c * NP_ + Z_ + z] + add * (-WL[z] * 1e-3f);
                    float qtn = QTw[c * Z_ + z] + DTSTEP_ * sqt;
                    QTw[c * Z_ + z] = qtn;
                    bool cond = qtn < EPSQ_;
                    total += qtn * WL[z];
                    lack  += cond ? WL[z] * EPSQ_ : 0.f;
                    valid += cond ? 0.f : qtn * WL[z];
                }
                float alpha = (total - lack) / valid;
                float q2 = 0.f, s2 = 0.f;
                for (int z = 0; z < Z_; ++z) {
                    float qtn = QTw[c * Z_ + z];
                    float qf  = (qtn < EPSQ_) ? EPSQ_ : qtn * alpha;
                    QTw[c * Z_ + z] = qf;
                    q2 += qf * WL[z];
                    s2 += SLw[c * Z_ + z] * WL[z];
                }
                qnn += (q2 - q0) * QNN_SCALE;
                snn += (s2 - s0) * SNN_SCALE;
            }
            wave_sync();
        } // inner steps

        // ---------------- outputs for this outer step ----------------------
        if (lane < 16) {
            const int c = lane, b = b0 + c;
            for (int z = 0; z < Z_; ++z) {
                out[(t + 1) * B_ * M_ + b * M_ + z]      = SLw[c * Z_ + z];
                out[(t + 1) * B_ * M_ + b * M_ + Z_ + z] = QTw[c * Z_ + z];
            }
            float* dg = out + PROG_CNT + (t * B_ + b) * 4;
            dg[0] = 0.0f;   // QLSF
            dg[1] = qnn;    // QNN
            dg[2] = 0.0f;   // SLSF
            dg[3] = snn;    // SNN
        }
        wave_sync();
    } // outer t
}

// ---------------------------------------------------------------------------
// Host launch
// ---------------------------------------------------------------------------
extern "C" void kernel_launch(void* const* d_in, const int* in_sizes, int n_in,
                              void* d_out, int out_size, void* d_ws, size_t ws_size,
                              hipStream_t stream) {
    const float* prog_sl  = (const float*)d_in[0];
    const float* prog_qt  = (const float*)d_in[1];
    const float* forc_sl  = (const float*)d_in[2];
    const float* forc_qt  = (const float*)d_in[3];
    const float* shf      = (const float*)d_in[4];
    const float* lhf      = (const float*)d_in[5];
    const float* solin    = (const float*)d_in[6];
    const float* w        = (const float*)d_in[7];
    const float* W1       = (const float*)d_in[8];
    const float* b1       = (const float*)d_in[9];
    const float* W2       = (const float*)d_in[10];
    const float* b2       = (const float*)d_in[11];
    const float* bn_gamma = (const float*)d_in[12];
    const float* bn_beta  = (const float*)d_in[13];
    float*       out      = (float*)d_out;
    float*       stats    = (float*)d_ws;   // (T-1) * 71 * 2 floats

    // Pass 1: hoisted BN statistics (forcing-only, loop-invariant per t)
    bn_stats_kernel<<<(T_ - 1) * NF_, 256, 0, stream>>>(
        shf, lhf, solin, forc_sl, forc_qt, bn_gamma, bn_beta, stats);

    // Pass 2: persistent WMMA time-stepper, 4 waves/block sharing weights
    forced_stepper_kernel<<<B_ / 64, 128, 0, stream>>>(
        prog_sl, prog_qt, forc_sl, forc_qt, shf, lhf, solin, w,
        W1, b1, W2, b2, stats, out);
}
